// DSSResNet_58325655880239
// MI455X (gfx1250) — compile-verified, matrix operand order checked by reading, not run on hardware
//
#include <hip/hip_runtime.h>
#include <hip/hip_bf16.h>
#include <math.h>

// ---------------- model constants ----------------
#define Bn   16
#define Hc   128
#define Ln   8192
#define L2n  16384
#define Nn   32
#define NRn  3
#define NLn  4
#define KW   896        // Cin*7 reduction for resnet conv
#define XTS  136        // padded f16 LDS row stride (272B: 16B-aligned, bank-spread)

typedef __attribute__((ext_vector_type(16))) _Float16 v16h;
typedef __attribute__((ext_vector_type(8)))  float    v8f;

struct c32 { float x, y; };
__device__ __forceinline__ c32 cmul(c32 a, c32 b) {
  return { a.x * b.x - a.y * b.y, a.x * b.y + a.y * b.x };
}

// WMMA 16-bit fragment K-index map: element e (0..15), lane-half lh (0/1)
__device__ __forceinline__ int kmap(int lh, int e) {
  return (e < 8) ? (lh * 8 + e) : (16 + lh * 8 + (e - 8));
}

// ---------------- encoder conv: (B,5,L) -> (B,H,L), K=15 same ----------------
__global__ __launch_bounds__(256) void k_encoder(const float* __restrict__ x,
                                                 const float* __restrict__ w,
                                                 const float* __restrict__ b,
                                                 float* __restrict__ out) {
  int idx = blockIdx.x * blockDim.x + threadIdx.x;
  if (idx >= Bn * Hc * Ln) return;
  int l  = idx % Ln;
  int co = (idx / Ln) % Hc;
  int bb = idx / (Ln * Hc);
  float acc = b[co];
  for (int ci = 0; ci < 5; ++ci) {
    const float* xr = x + ((size_t)bb * 5 + ci) * Ln;
    const float* wr = w + ((size_t)co * 5 + ci) * 15;
#pragma unroll
    for (int k = 0; k < 15; ++k) {
      int p = l + k - 7;
      float xv = (p >= 0 && p < Ln) ? xr[p] : 0.f;
      acc = fmaf(xv, wr[k], acc);
    }
  }
  out[idx] = acc;
}

// ---------------- weight packing (BN fold + f16 convert) ----------------
__global__ __launch_bounds__(256) void k_pack(const float* rw1, const float* rb1,
                                              const float* rg1, const float* rbe1,
                                              const float* rw2, const float* rb2,
                                              const float* rg2, const float* rbe2,
                                              const float* out_w,
                                              _Float16* WP, float* BF, _Float16* OWP) {
  const float inv = rsqrtf(1.f + 1e-5f);
  int t = blockIdx.x * blockDim.x + threadIdx.x;
  const int total_w = NRn * 2 * Hc * 7 * Hc; // dst: ((blk*2+s)*H+co)*896 + k*128 + ci
  if (t < total_w) {
    int ci = t % Hc;
    int k  = (t / Hc) % 7;
    int co = (t / (Hc * 7)) % Hc;
    int s  = (t / (Hc * 7 * Hc)) % 2;
    int blk = t / (Hc * 7 * Hc * 2);
    const float* w = s ? rw2 : rw1;
    const float* g = s ? rg2 : rg1;
    float scale = g[blk * Hc + co] * inv;
    WP[t] = (_Float16)(w[(((size_t)blk * Hc + co) * Hc + ci) * 7 + k] * scale);
    return;
  }
  int t2 = t - total_w;
  if (t2 < NRn * 2 * Hc) {
    int co = t2 % Hc;
    int s  = (t2 / Hc) % 2;
    int blk = t2 / (2 * Hc);
    const float* bb = s ? rb2 : rb1;
    const float* g  = s ? rg2 : rg1;
    const float* be = s ? rbe2 : rbe1;
    BF[t2] = bb[blk * Hc + co] * (g[blk * Hc + co] * inv) + be[blk * Hc + co];
    return;
  }
  int t3 = t2 - NRn * 2 * Hc;
  if (t3 < NLn * 256 * Hc) OWP[t3] = (_Float16)out_w[t3];
}

// ---------------- resnet conv via WMMA implicit GEMM ----------------
// Out[co, pos] = relu( sum_{ci,k} Wp[co][k*128+ci] * Xin[ci][pos+k-3] + bias[co] (+ Res) )
// Input tile staged with gfx1250 async global->LDS DMA, converted to f16 in LDS,
// epilogue transposed through LDS for coalesced residual-read + store.
template <bool HAS_RES>
__global__ __launch_bounds__(256) void k_resconv(const float* __restrict__ Xin,
                                                 const _Float16* __restrict__ Wp,
                                                 const float* __restrict__ bias,
                                                 const float* __restrict__ Res,
                                                 float* __restrict__ Out) {
  // staging/transpose buffer (aliased): xs = float[128][72] ; zt = float[128][66]
  __shared__ __align__(16) char sbuf[36864];
  __shared__ __align__(16) _Float16 xt[70 * XTS]; // [pp][ci] padded rows, 19KB
  float* xs = (float*)sbuf;
  float* zt = (float*)sbuf;

  const int nblk_l = Ln / 64;
  int b = blockIdx.x / nblk_l;
  int pbase = (blockIdx.x % nblk_l) * 64;
  int tid = threadIdx.x;

  // ---- async DMA fill: global rows -> xs[ci][pp] (pp contiguous) ----
  for (int i = tid; i < 70 * Hc; i += 256) {
    int ci = i / 70;
    int pp = i - ci * 70;
    int gp = pbase - 3 + pp;
    float* dst = &xs[ci * 72 + pp];
    if (gp >= 0 && gp < Ln) {
      unsigned int ldsa = (unsigned int)(unsigned long long)(uintptr_t)dst; // LDS byte offset
      unsigned long long ga =
          (unsigned long long)(uintptr_t)(Xin + ((size_t)b * Hc + ci) * Ln + gp);
      asm volatile("global_load_async_to_lds_b32 %0, %1, off"
                   :: "v"(ldsa), "v"(ga) : "memory");
    } else {
      *dst = 0.f;
    }
  }
  asm volatile("s_wait_asynccnt 0x0" ::: "memory");
  __syncthreads();

  // ---- in-LDS transpose + f16 convert: xs[ci][pp] -> xt[pp][ci] ----
  for (int i = tid; i < 70 * Hc; i += 256) {
    int pp = i / Hc, ci = i - pp * Hc;
    xt[pp * XTS + ci] = (_Float16)xs[ci * 72 + pp];
  }
  __syncthreads();

  int wave = tid >> 5, lane = tid & 31;
  int lh = lane >> 4, ln = lane & 15;
  int mt = wave * 16;            // 8 waves cover 128 out channels
  v8f accs[4] = {};              // 4 N-tiles of 16 positions

  for (int c = 0; c < 28; ++c) { // K = 896 = 28 chunks of 32
    int kconv = c >> 2;
    int cbase = (c & 3) * 32;
    v16h a;
    const _Float16* wrow = Wp + (size_t)(mt + ln) * KW + c * 32;
    __builtin_prefetch(wrow + 32, 0, 1);    // global_prefetch_b8 (next chunk)
#pragma unroll
    for (int e = 0; e < 16; ++e) a[e] = wrow[kmap(lh, e)];
#pragma unroll
    for (int nt = 0; nt < 4; ++nt) {
      int pp = nt * 16 + ln + kconv;
      v16h bf;
#pragma unroll
      for (int e = 0; e < 16; ++e) bf[e] = xt[pp * XTS + cbase + kmap(lh, e)];
      accs[nt] = __builtin_amdgcn_wmma_f32_16x16x32_f16(
          false, a, false, bf, (short)0, accs[nt], false, false);
    }
  }

  // ---- epilogue: transpose through LDS, then coalesced bias/res/relu/store ----
#pragma unroll
  for (int nt = 0; nt < 4; ++nt) {
#pragma unroll
    for (int r = 0; r < 8; ++r) {
      int m = mt + r + 8 * lh;
      int col = nt * 16 + ln;
      zt[m * 66 + col] = accs[nt][r];
    }
  }
  __syncthreads();
  for (int i = tid; i < Hc * 64; i += 256) {
    int m = i >> 6, col = i & 63;
    size_t gi = ((size_t)b * Hc + m) * Ln + pbase + col;
    float v = zt[m * 66 + col] + bias[m];
    if (HAS_RES) v += Res[gi];
    Out[gi] = fmaxf(v, 0.f);
  }
}

// ---------------- DSS prep: dtL and Wc (complex) ----------------
__global__ __launch_bounds__(256) void k_dssprep(const float* lam_re, const float* lam_im,
                                                 const float* log_dt, const float* W_re,
                                                 const float* W_im, int layer,
                                                 c32* Wc, c32* dtL) {
  int t = blockIdx.x * blockDim.x + threadIdx.x;
  if (t >= Hc * Nn) return;
  int h = t / Nn, n = t - h * Nn;
  float lre = -__expf(lam_re[layer * Nn + n]);
  float lim = lam_im[layer * Nn + n];
  float dt  = __expf(log_dt[layer * Hc + h]);
  c32 dl{dt * lre, dt * lim};
  dtL[t] = dl;
  float er = __expf(dl.x);
  float sn, cs; __sincosf(dl.y, &sn, &cs);
  c32 ed{er * cs - 1.f, er * sn};
  float den = lre * lre + lim * lim;
  c32 f{(ed.x * lre + ed.y * lim) / den, (ed.y * lre - ed.x * lim) / den};
  for (int cc = 0; cc < 2; ++cc) {
    int wi = (((layer * 2 + cc) * Hc) + h) * Nn + n;
    c32 w{W_re[wi], W_im[wi]};
    Wc[(cc * Hc + h) * Nn + n] = cmul(w, f);
  }
}

// ---------------- build bidirectional time-domain kernel k (H, 2L) ----------------
__global__ __launch_bounds__(256) void k_buildk(const c32* __restrict__ Wc,
                                                const c32* __restrict__ dtL,
                                                float* __restrict__ kt) {
  int idx = blockIdx.x * blockDim.x + threadIdx.x;
  if (idx >= Hc * L2n) return;
  int h = idx / L2n, l2 = idx - h * L2n;
  int c, l;
  if (l2 < Ln) { c = 0; l = l2; } else { c = 1; l = Ln - 1 - (l2 - Ln); }
  float acc = 0.f;
  const c32* dr = dtL + h * Nn;
  const c32* wr = Wc + (c * Hc + h) * Nn;
  float fl = (float)l;
#pragma unroll 4
  for (int n = 0; n < Nn; ++n) {
    c32 dl = dr[n];
    float mag = __expf(dl.x * fl);
    float sn, cs; __sincosf(dl.y * fl, &sn, &cs);
    c32 w = wr[n];
    acc = fmaf(w.x, mag * cs, acc);
    acc = fmaf(-w.y, mag * sn, acc);
  }
  kt[idx] = acc;
}

// ---------------- in-LDS radix-2 FFT (16K points, 128KB) ----------------
__device__ void fft_lds(c32* s, int logn, bool inverse, int tid, int nthr) {
  const int n = 1 << logn;
  for (int i = tid; i < n; i += nthr) {
    int j = (int)(__brev((unsigned)i) >> (32 - logn));
    if (j > i) { c32 t = s[i]; s[i] = s[j]; s[j] = t; }
  }
  __syncthreads();
  for (int len = 2; len <= n; len <<= 1) {
    const int half = len >> 1;
    const float ang0 = (inverse ? 6.28318530717958647692f : -6.28318530717958647692f) / (float)len;
    for (int bi = tid; bi < (n >> 1); bi += nthr) {
      int grp = bi / half;
      int k   = bi - grp * half;
      int i0  = grp * len + k;
      int i1  = i0 + half;
      float sn, cs; __sincosf(ang0 * (float)k, &sn, &cs);
      c32 w{cs, sn};
      c32 u = s[i0];
      c32 v = cmul(s[i1], w);
      s[i0] = { u.x + v.x, u.y + v.y };
      s[i1] = { u.x - v.x, u.y - v.y };
    }
    __syncthreads();
  }
}

// FFT of the time-domain kernel rows -> full complex spectrum (H, 2L)
__global__ __launch_bounds__(256) void k_fftk(const float* __restrict__ kt, c32* __restrict__ kf) {
  extern __shared__ c32 smk[];
  int h = blockIdx.x, tid = threadIdx.x;
  const float* row = kt + (size_t)h * L2n;
  for (int i = tid; i < L2n; i += 256) smk[i] = { row[i], 0.f };
  __syncthreads();
  fft_lds(smk, 14, false, tid, 256);
  c32* krow = kf + (size_t)h * L2n;
  for (int i = tid; i < L2n; i += 256) krow[i] = smk[i];
}

// fused: zero-pad FFT(x) * k_f -> IFFT -> +D*x -> GELU, per (b,h) row
__global__ __launch_bounds__(256) void k_dssfused(const float* __restrict__ X,
                                                  const c32* __restrict__ kf,
                                                  const float* __restrict__ Dp,
                                                  float* __restrict__ Y) {
  extern __shared__ c32 smx[];
  int bh = blockIdx.x;
  int h = bh % Hc;
  int tid = threadIdx.x;
  const float* xr = X + (size_t)bh * Ln;
  for (int i = tid; i < L2n; i += 256) smx[i] = { (i < Ln) ? xr[i] : 0.f, 0.f };
  __syncthreads();
  fft_lds(smx, 14, false, tid, 256);
  const c32* krow = kf + (size_t)h * L2n;
  for (int i = tid; i < L2n; i += 256) smx[i] = cmul(smx[i], krow[i]);
  __syncthreads();
  fft_lds(smx, 14, true, tid, 256);
  const float invn = 1.0f / (float)L2n;
  const float dv = Dp[h];
  float* yr = Y + (size_t)bh * Ln;
  for (int i = tid; i < Ln; i += 256) {
    float v = smx[i].x * invn + dv * xr[i];
    yr[i] = 0.5f * v * (1.0f + erff(v * 0.70710678118654752f)); // exact GELU
  }
}

// ---------------- out_w GEMM (WMMA) + GLU + residual + channel LayerNorm ----------------
__global__ __launch_bounds__(256) void k_glu(const float* __restrict__ Yg,
                                             const _Float16* __restrict__ OwP,
                                             const float* __restrict__ ob,
                                             const float* __restrict__ Xin,
                                             const float* __restrict__ lng,
                                             const float* __restrict__ lnb,
                                             float* __restrict__ Xout) {
  __shared__ __align__(16) _Float16 yt[16 * XTS]; // [p][h] padded rows (4.4KB)
  __shared__ float zt[256 * 17];                  // [m][p] padded (17.4KB)
  __shared__ float redm[256], redv[256];
  __shared__ float mu_s[16], rs_s[16];
  const int nblk_l = Ln / 16;
  int b = blockIdx.x / nblk_l;
  int pbase = (blockIdx.x % nblk_l) * 16;
  int tid = threadIdx.x;

  // coalesced global read (p fastest), transposed scatter into [p][h]
  for (int i = tid; i < Hc * 16; i += 256) {
    int p = i & 15, hh = i >> 4;
    yt[p * XTS + hh] = (_Float16)Yg[((size_t)b * Hc + hh) * Ln + pbase + p];
  }
  __syncthreads();

  int wave = tid >> 5, lane = tid & 31, lh = lane >> 4, ln = lane & 15;
  v8f acc[2] = {};
#pragma unroll
  for (int c = 0; c < 4; ++c) {     // K = 128
    v16h bf;
#pragma unroll
    for (int e = 0; e < 16; ++e) bf[e] = yt[ln * XTS + c * 32 + kmap(lh, e)];
#pragma unroll
    for (int mi = 0; mi < 2; ++mi) { // 8 waves * 32 rows = 256 out rows
      int mt = wave * 32 + mi * 16;
      v16h a;
      const _Float16* wrow = OwP + (size_t)(mt + ln) * Hc + c * 32;
#pragma unroll
      for (int e = 0; e < 16; ++e) a[e] = wrow[kmap(lh, e)];
      acc[mi] = __builtin_amdgcn_wmma_f32_16x16x32_f16(
          false, a, false, bf, (short)0, acc[mi], false, false);
    }
  }
#pragma unroll
  for (int mi = 0; mi < 2; ++mi)
#pragma unroll
    for (int r = 0; r < 8; ++r) {
      int m = wave * 32 + mi * 16 + r + 8 * lh;
      zt[m * 17 + ln] = acc[mi][r] + ob[m];
    }
  __syncthreads();

  // GLU + residual into zt lower half
  for (int i = tid; i < Hc * 16; i += 256) {
    int m = i / 16, p = i - m * 16;
    float av = zt[m * 17 + p];
    float gv = zt[(m + Hc) * 17 + p];
    float sg = 1.f / (1.f + __expf(-gv));
    zt[m * 17 + p] = av * sg + Xin[((size_t)b * Hc + m) * Ln + pbase + p];
  }
  __syncthreads();

  { // LN partial sums: 16 groups x 8 channels per position
    int p = tid & 15, grp = tid >> 4;
    float sm = 0.f, sq = 0.f;
#pragma unroll
    for (int j = 0; j < 8; ++j) {
      float v = zt[(grp * 8 + j) * 17 + p];
      sm += v; sq += v * v;
    }
    redm[tid] = sm; redv[tid] = sq;
  }
  __syncthreads();
  if (tid < 16) {
    float sm = 0.f, sq = 0.f;
    for (int g = 0; g < 16; ++g) { sm += redm[g * 16 + tid]; sq += redv[g * 16 + tid]; }
    float mu = sm * (1.0f / Hc);
    float var = sq * (1.0f / Hc) - mu * mu;
    mu_s[tid] = mu;
    rs_s[tid] = rsqrtf(var + 1e-5f);
  }
  __syncthreads();
  for (int i = tid; i < Hc * 16; i += 256) {
    int m = i / 16, p = i - m * 16;
    float v = (zt[m * 17 + p] - mu_s[p]) * rs_s[p] * lng[m] + lnb[m];
    Xout[((size_t)b * Hc + m) * Ln + pbase + p] = v;
  }
}

// ---------------- decoder conv: (B,H,L) -> (B,5,L), K=15 same ----------------
__global__ __launch_bounds__(256) void k_decoder(const float* __restrict__ X,
                                                 const float* __restrict__ dw,
                                                 const float* __restrict__ db,
                                                 float* __restrict__ out) {
  int idx = blockIdx.x * blockDim.x + threadIdx.x;
  if (idx >= Bn * 5 * Ln) return;
  int l  = idx % Ln;
  int co = (idx / Ln) % 5;
  int b  = idx / (5 * Ln);
  float acc = db[co];
  for (int ci = 0; ci < Hc; ++ci) {
    const float* xr = X + ((size_t)b * Hc + ci) * Ln;
    const float* wr = dw + ((size_t)co * Hc + ci) * 15;
#pragma unroll
    for (int k = 0; k < 15; ++k) {
      int p = l + k - 7;
      float xv = (p >= 0 && p < Ln) ? xr[p] : 0.f;
      acc = fmaf(xv, wr[k], acc);
    }
  }
  out[idx] = acc;
}

// ---------------- host orchestration ----------------
extern "C" void kernel_launch(void* const* d_in, const int* in_sizes, int n_in,
                              void* d_out, int out_size, void* d_ws, size_t ws_size,
                              hipStream_t stream) {
  (void)in_sizes; (void)n_in; (void)out_size; (void)ws_size;
  const float* x      = (const float*)d_in[0];
  const float* enc_w  = (const float*)d_in[1];
  const float* enc_b  = (const float*)d_in[2];
  const float* rn_w1  = (const float*)d_in[3];
  const float* rn_b1  = (const float*)d_in[4];
  const float* rn_g1  = (const float*)d_in[5];
  const float* rn_be1 = (const float*)d_in[6];
  const float* rn_w2  = (const float*)d_in[7];
  const float* rn_b2  = (const float*)d_in[8];
  const float* rn_g2  = (const float*)d_in[9];
  const float* rn_be2 = (const float*)d_in[10];
  const float* lam_re = (const float*)d_in[11];
  const float* lam_im = (const float*)d_in[12];
  const float* log_dt = (const float*)d_in[13];
  const float* W_re   = (const float*)d_in[14];
  const float* W_im   = (const float*)d_in[15];
  const float* Dp     = (const float*)d_in[16];
  const float* out_w  = (const float*)d_in[17];
  const float* out_b  = (const float*)d_in[18];
  const float* ln_g   = (const float*)d_in[19];
  const float* ln_b   = (const float*)d_in[20];
  const float* dec_w  = (const float*)d_in[21];
  const float* dec_b  = (const float*)d_in[22];

  char* ws = (char*)d_ws;
  const size_t ACT = (size_t)Bn * Hc * Ln * sizeof(float);      // 64MB
  float* XA = (float*)(ws);
  float* XB = (float*)(ws + ACT);
  float* Yb = (float*)(ws + 2 * ACT);
  float* KT = (float*)(ws + 3 * ACT);                            // H*2L f32 = 8MB
  c32*   KF = (c32*)  (ws + 3 * ACT + (size_t)Hc * L2n * 4);     // H*2L c32 = 16MB
  char*  p  = ws + 3 * ACT + (size_t)Hc * L2n * 4 + (size_t)Hc * L2n * 8;
  c32* WC  = (c32*)p;                 p += (size_t)2 * Hc * Nn * sizeof(c32);
  c32* DTL = (c32*)p;                 p += (size_t)Hc * Nn * sizeof(c32);
  _Float16* WP  = (_Float16*)p;       p += (size_t)NRn * 2 * Hc * KW * sizeof(_Float16);
  _Float16* OWP = (_Float16*)p;       p += (size_t)NLn * 256 * Hc * sizeof(_Float16);
  float* BF = (float*)p;

  // pack weights (BN fold, f16 convert)
  {
    int total = NRn * 2 * Hc * 7 * Hc + NRn * 2 * Hc + NLn * 256 * Hc;
    k_pack<<<(total + 255) / 256, 256, 0, stream>>>(rn_w1, rn_b1, rn_g1, rn_be1,
                                                    rn_w2, rn_b2, rn_g2, rn_be2,
                                                    out_w, WP, BF, OWP);
  }

  // encoder
  k_encoder<<<(Bn * Hc * Ln) / 256, 256, 0, stream>>>(x, enc_w, enc_b, XA);

  float* cur = XA;
  float* oth = XB;
  // resnet blocks
  for (int i = 0; i < NRn; ++i) {
    const _Float16* w1 = WP + (size_t)(i * 2 + 0) * Hc * KW;
    const _Float16* w2 = WP + (size_t)(i * 2 + 1) * Hc * KW;
    const float* b1 = BF + (i * 2 + 0) * Hc;
    const float* b2 = BF + (i * 2 + 1) * Hc;
    k_resconv<false><<<Bn * (Ln / 64), 256, 0, stream>>>(cur, w1, b1, nullptr, Yb);
    k_resconv<true><<<Bn * (Ln / 64), 256, 0, stream>>>(Yb, w2, b2, cur, oth);
    float* t = cur; cur = oth; oth = t;
  }

  // DSS blocks
  for (int l = 0; l < NLn; ++l) {
    k_dssprep<<<(Hc * Nn + 255) / 256, 256, 0, stream>>>(lam_re, lam_im, log_dt,
                                                         W_re, W_im, l, WC, DTL);
    k_buildk<<<(Hc * L2n) / 256, 256, 0, stream>>>(WC, DTL, KT);
    k_fftk<<<Hc, 256, L2n * sizeof(c32), stream>>>(KT, KF);
    k_dssfused<<<Bn * Hc, 256, L2n * sizeof(c32), stream>>>(cur, KF, Dp + l * Hc, Yb);
    k_glu<<<Bn * (Ln / 16), 256, 0, stream>>>(Yb, OWP + (size_t)l * 256 * Hc,
                                              out_b + l * 256, cur,
                                              ln_g + l * Hc, ln_b + l * Hc, oth);
    float* t = cur; cur = oth; oth = t;
  }

  // decoder
  k_decoder<<<(Bn * 5 * Ln + 255) / 256, 256, 0, stream>>>(cur, dec_w, dec_b, (float*)d_out);
}
